// HeteroGNN_89163521065346
// MI455X (gfx1250) — compile-verified, matrix operand order checked by reading, not run on hardware
//
#include <hip/hip_runtime.h>
#include <hip/hip_bf16.h>

// ---------------------------------------------------------------------------
// HeteroGNN (2-layer bipartite GraphSAGE) for MI455X / gfx1250, wave32 + WMMA.
//
// Pipeline:
//   cnt_flow = seg_count(dst_hf), cnt_host = seg_count(dst_fh)
//   agg_flow = seg_sum(x_host[src_hf] -> dst_hf, D=64)
//   agg_host = seg_sum(x_flow[src_fh] -> dst_fh, D=64)
//   h_flow = lrelu((agg_flow/cnt) @ W0_hf_l + b + x_flow @ W0_hf_r)   [200000,128]
//   h_host = lrelu((agg_host/cnt) @ W0_fh_l + b + x_host @ W0_fh_r)   [20000,128]
//   agg_flow = seg_sum(h_host[src_hf] -> dst_hf, D=128)
//   g_flow = lrelu((agg_flow/cnt) @ W1_hf_l + b + h_flow @ W1_hf_r)   [200000,128]
//   out    = g_flow @ W_out + b_out                                    [200000,32]
// (g_host is dead in the reference and skipped.)
//
// GEMMs use v_wmma_f32_16x16x32_f16: fp32 data converted to f16 operands,
// fp32 accumulation. Weights staged in LDS pre-swizzled to fragment layout.
// ---------------------------------------------------------------------------

typedef __attribute__((ext_vector_type(16))) _Float16 v16h;
typedef __attribute__((ext_vector_type(8)))  float    v8f;

#define N_HOST_C 20000
#define N_FLOW_C 200000
#define E_C      600000
#define D_IN_C   64
#define D_H_C    128
#define D_OUT_C  32

// ---------------- utility kernels ----------------

__global__ __launch_bounds__(256) void zero_kernel(float* __restrict__ p, long long n) {
    long long i = (long long)blockIdx.x * blockDim.x + threadIdx.x;
    long long stride = (long long)gridDim.x * blockDim.x;
    for (; i < n; i += stride) p[i] = 0.0f;
}

__global__ __launch_bounds__(256) void count_kernel(const int* __restrict__ dst,
                                                    float* __restrict__ cnt, int E) {
    int e = blockIdx.x * blockDim.x + threadIdx.x;
    if (e < E) atomicAdd(&cnt[dst[e]], 1.0f);
}

// one thread per (edge, 4-float chunk); consecutive threads read consecutive
// floats of the same source row -> coalesced gather, fp32 atomic scatter.
__global__ __launch_bounds__(256) void scatter_add_kernel(const float* __restrict__ x,
                                                          const int* __restrict__ src,
                                                          const int* __restrict__ dst,
                                                          float* __restrict__ agg,
                                                          int E, int D) {
    long long idx = (long long)blockIdx.x * blockDim.x + threadIdx.x;
    const int vpe = D >> 2;
    if (idx >= (long long)E * vpe) return;
    int e = (int)(idx / vpe);
    int c = (int)(idx - (long long)e * vpe);
    int s = src[e];
    int d = dst[e];
    const float4 v = *(const float4*)(x + (size_t)s * D + (size_t)c * 4);
    float* o = agg + (size_t)d * D + (size_t)c * 4;
    atomicAdd(o + 0, v.x);
    atomicAdd(o + 1, v.y);
    atomicAdd(o + 2, v.z);
    atomicAdd(o + 3, v.w);
}

// ---------------- WMMA fragment helpers ----------------
// 16-bit A-matrix 16x32 layout (CDNA5 ISA):
//   lanes 0-15: M=lane,   K in {0..7} (v[0..7]) and {16..23} (v[8..15])
//   lanes16-31: M=lane-16,K in {8..15}          and {24..31}
// slot pair j (0..7): koff = j<4 ? 2j : 16+2(j-4); kbase = (lane&16)?8:0.
// B (32x16) mirrors A with lane -> column N. C/D 16x16 f32: VGPR r holds
// row M = r + 8*(lane>>4), col N = lane&15.

__device__ __forceinline__ int frag_koff(int j) {
    return (j < 4) ? (2 * j) : (16 + 2 * (j - 4));
}

// Build A fragment from an fp32 row (row-major, stride K), with scale fused.
__device__ __forceinline__ v16h load_a_frag(const float* __restrict__ rowp,
                                            int kbase, float scale) {
    v16h a;
#pragma unroll
    for (int j = 0; j < 8; ++j) {
        const float2 p = *(const float2*)(rowp + kbase + frag_koff(j));
        a[2 * j + 0] = (_Float16)(p.x * scale);
        a[2 * j + 1] = (_Float16)(p.y * scale);
    }
    return a;
}

// Stage fp32 weight matrix W[K][N] (row-major, K = KSTEPS*32, N = NT*16) into
// LDS as f16, pre-swizzled into per-(tile,kstep) 32-lane x 16-elem fragments so
// compute lanes read contiguous 32B (two ds_load_b128).
template <int KSTEPS, int NT>
__device__ __forceinline__ void stage_w(const float* __restrict__ W, _Float16* sW) {
    const int N = NT * 16;
    const int total = KSTEPS * 32 * N;  // K*N f16 elements
    for (int e = threadIdx.x; e < total; e += blockDim.x) {
        const int frag = e >> 9;       // 512 f16 per fragment
        const int r    = e & 511;
        const int lane = r >> 4;
        const int j16  = r & 15;
        const int t = frag / KSTEPS;
        const int s = frag - t * KSTEPS;
        const int n = t * 16 + (lane & 15);
        const int k = s * 32 + ((lane & 16) ? 8 : 0) + frag_koff(j16 >> 1) + (j16 & 1);
        sW[e] = (_Float16)W[k * N + n];
    }
}

// ---------------- fused SAGE layer: out = lrelu(agg/cnt @ Wl + b + xdst @ Wr)
// K = KSTEPS*32 in {64,128}; N = 128 fixed. 256 threads = 8 waves; each wave
// owns a 16-row M tile -> 128 rows per block.
template <int KSTEPS>
__global__ __launch_bounds__(256) void sage_kernel(const float* __restrict__ agg,
                                                   const float* __restrict__ cnt,
                                                   const float* __restrict__ xdst,
                                                   const float* __restrict__ Wl,
                                                   const float* __restrict__ Wr,
                                                   const float* __restrict__ bias,
                                                   float* __restrict__ out, int M) {
    constexpr int K = KSTEPS * 32;
    __shared__ __align__(32) _Float16 sW[K * D_H_C];

    const int lane = threadIdx.x & 31;
    const int wave = threadIdx.x >> 5;
    const int row0 = blockIdx.x * 128 + wave * 16;
    const bool active = (row0 < M);               // wave-uniform; M % 16 == 0
    const int m = row0 + (lane & 15);
    const int kb_lane = (lane & 16) ? 8 : 0;

    v8f acc[8] = {};

    // ---- pass 1: (agg / cnt) @ Wl ----
    stage_w<KSTEPS, 8>(Wl, sW);
    __syncthreads();
    if (active) {
        const float inv = 1.0f / fmaxf(cnt[m], 1.0f);
        const float* rowp = agg + (size_t)m * K;
#pragma unroll
        for (int s = 0; s < KSTEPS; ++s) {
            const v16h a = load_a_frag(rowp, s * 32 + kb_lane, inv);
#pragma unroll
            for (int t = 0; t < 8; ++t) {
                const v16h* bp = (const v16h*)(sW + ((size_t)(t * KSTEPS + s) * 512 + lane * 16));
                acc[t] = __builtin_amdgcn_wmma_f32_16x16x32_f16(
                    false, a, false, *bp, (short)0, acc[t], false, false);
            }
        }
    }
    __syncthreads();

    // ---- pass 2: + xdst @ Wr ----
    stage_w<KSTEPS, 8>(Wr, sW);
    __syncthreads();
    if (active) {
        const float* rowp = xdst + (size_t)m * K;
#pragma unroll
        for (int s = 0; s < KSTEPS; ++s) {
            const v16h a = load_a_frag(rowp, s * 32 + kb_lane, 1.0f);
#pragma unroll
            for (int t = 0; t < 8; ++t) {
                const v16h* bp = (const v16h*)(sW + ((size_t)(t * KSTEPS + s) * 512 + lane * 16));
                acc[t] = __builtin_amdgcn_wmma_f32_16x16x32_f16(
                    false, a, false, *bp, (short)0, acc[t], false, false);
            }
        }
        // epilogue: bias + leaky relu, C/D layout -> global
        const int mbase = row0 + ((lane & 16) ? 8 : 0);
        const int ncol = lane & 15;
#pragma unroll
        for (int t = 0; t < 8; ++t) {
            const int n = t * 16 + ncol;
            const float bv = bias[n];
#pragma unroll
            for (int r = 0; r < 8; ++r) {
                float v = acc[t][r] + bv;
                v = (v > 0.0f) ? v : 0.01f * v;
                out[(size_t)(mbase + r) * D_H_C + n] = v;
            }
        }
    }
}

// ---------------- head: out = g @ W_out + b_out, K=128, N=32 ----------------
__global__ __launch_bounds__(256) void head_kernel(const float* __restrict__ g,
                                                   const float* __restrict__ W,
                                                   const float* __restrict__ bias,
                                                   float* __restrict__ out, int M) {
    __shared__ __align__(32) _Float16 sW[D_H_C * D_OUT_C];

    const int lane = threadIdx.x & 31;
    const int wave = threadIdx.x >> 5;
    const int row0 = blockIdx.x * 128 + wave * 16;
    const bool active = (row0 < M);
    const int m = row0 + (lane & 15);
    const int kb_lane = (lane & 16) ? 8 : 0;

    stage_w<4, 2>(W, sW);
    __syncthreads();
    if (!active) return;

    v8f acc[2] = {};
    const float* rowp = g + (size_t)m * D_H_C;
#pragma unroll
    for (int s = 0; s < 4; ++s) {
        const v16h a = load_a_frag(rowp, s * 32 + kb_lane, 1.0f);
#pragma unroll
        for (int t = 0; t < 2; ++t) {
            const v16h* bp = (const v16h*)(sW + ((size_t)(t * 4 + s) * 512 + lane * 16));
            acc[t] = __builtin_amdgcn_wmma_f32_16x16x32_f16(
                false, a, false, *bp, (short)0, acc[t], false, false);
        }
    }
    const int mbase = row0 + ((lane & 16) ? 8 : 0);
    const int ncol = lane & 15;
#pragma unroll
    for (int t = 0; t < 2; ++t) {
        const int n = t * 16 + ncol;
        const float bv = bias[n];
#pragma unroll
        for (int r = 0; r < 8; ++r) {
            out[(size_t)(mbase + r) * D_OUT_C + n] = acc[t][r] + bv;
        }
    }
}

// ---------------- launch ----------------

extern "C" void kernel_launch(void* const* d_in, const int* in_sizes, int n_in,
                              void* d_out, int out_size, void* d_ws, size_t ws_size,
                              hipStream_t stream) {
    (void)in_sizes; (void)n_in; (void)out_size; (void)ws_size;

    const float* x_host  = (const float*)d_in[0];
    const float* x_flow  = (const float*)d_in[1];
    const int*   src_hf  = (const int*)d_in[2];
    const int*   dst_hf  = (const int*)d_in[3];
    const int*   src_fh  = (const int*)d_in[4];
    const int*   dst_fh  = (const int*)d_in[5];
    const float* W0_hf_l = (const float*)d_in[6];
    const float* W0_hf_r = (const float*)d_in[7];
    const float* b0_hf   = (const float*)d_in[8];
    const float* W0_fh_l = (const float*)d_in[9];
    const float* W0_fh_r = (const float*)d_in[10];
    const float* b0_fh   = (const float*)d_in[11];
    const float* W1_hf_l = (const float*)d_in[12];
    const float* W1_hf_r = (const float*)d_in[13];
    const float* b1_hf   = (const float*)d_in[14];
    const float* W_out   = (const float*)d_in[18];
    const float* b_out   = (const float*)d_in[19];
    float* out = (float*)d_out;

    // workspace layout (fp32 elements)
    float* ws = (float*)d_ws;
    float* agg_flow = ws;                                    // 200000*128
    float* agg_host = agg_flow + (size_t)N_FLOW_C * D_H_C;   // 20000*128
    float* h_flow   = agg_host + (size_t)N_HOST_C * D_H_C;   // 200000*128
    float* h_host   = h_flow   + (size_t)N_FLOW_C * D_H_C;   // 20000*128
    float* g_flow   = h_host   + (size_t)N_HOST_C * D_H_C;   // 200000*128
    float* cnt_flow = g_flow   + (size_t)N_FLOW_C * D_H_C;   // 200000
    float* cnt_host = cnt_flow + (size_t)N_FLOW_C;           // 20000

    const int ZB = 2048;
    // zero accumulators + counts
    zero_kernel<<<ZB, 256, 0, stream>>>(agg_flow, (long long)N_FLOW_C * D_IN_C);
    zero_kernel<<<ZB, 256, 0, stream>>>(agg_host, (long long)N_HOST_C * D_IN_C);
    zero_kernel<<<ZB, 256, 0, stream>>>(cnt_flow, (long long)N_FLOW_C);
    zero_kernel<<<ZB, 256, 0, stream>>>(cnt_host, (long long)N_HOST_C);

    // degree counts (same edge lists both layers)
    count_kernel<<<(E_C + 255) / 256, 256, 0, stream>>>(dst_hf, cnt_flow, E_C);
    count_kernel<<<(E_C + 255) / 256, 256, 0, stream>>>(dst_fh, cnt_host, E_C);

    // layer 0 aggregation (D=64)
    {
        long long tot = (long long)E_C * (D_IN_C / 4);
        int blocks = (int)((tot + 255) / 256);
        scatter_add_kernel<<<blocks, 256, 0, stream>>>(x_host, src_hf, dst_hf, agg_flow, E_C, D_IN_C);
        scatter_add_kernel<<<blocks, 256, 0, stream>>>(x_flow, src_fh, dst_fh, agg_host, E_C, D_IN_C);
    }

    // layer 0 fused SAGE GEMMs (K=64)
    sage_kernel<2><<<(N_FLOW_C + 127) / 128, 256, 0, stream>>>(
        agg_flow, cnt_flow, x_flow, W0_hf_l, W0_hf_r, b0_hf, h_flow, N_FLOW_C);
    sage_kernel<2><<<(N_HOST_C + 127) / 128, 256, 0, stream>>>(
        agg_host, cnt_host, x_host, W0_fh_l, W0_fh_r, b0_fh, h_host, N_HOST_C);

    // layer 1 flow aggregation (D=128); g_host branch is dead in the reference
    zero_kernel<<<ZB, 256, 0, stream>>>(agg_flow, (long long)N_FLOW_C * D_H_C);
    {
        long long tot = (long long)E_C * (D_H_C / 4);
        int blocks = (int)((tot + 255) / 256);
        scatter_add_kernel<<<blocks, 256, 0, stream>>>(h_host, src_hf, dst_hf, agg_flow, E_C, D_H_C);
    }

    // layer 1 fused SAGE GEMM (K=128)
    sage_kernel<4><<<(N_FLOW_C + 127) / 128, 256, 0, stream>>>(
        agg_flow, cnt_flow, h_flow, W1_hf_l, W1_hf_r, b1_hf, g_flow, N_FLOW_C);

    // output head
    head_kernel<<<(N_FLOW_C + 127) / 128, 256, 0, stream>>>(g_flow, W_out, b_out, out, N_FLOW_C);
}
